// ShadowDropTransformer_15058155339884
// MI455X (gfx1250) — matrix operand order repaired and checked by reading, not compiled, not run on hardware
//
#include <hip/hip_runtime.h>
#include <hip/hip_bf16.h>

typedef _Float16 h4  __attribute__((ext_vector_type(4)));
typedef _Float16 h8  __attribute__((ext_vector_type(8)));
typedef _Float16 v16h __attribute__((ext_vector_type(16)));
typedef float    v8f  __attribute__((ext_vector_type(8)));

constexpr int Dm = 1024, Hh = 16, FFf = 2752, Ll = 4, Vv = 49152, GSz = 128, DHd = 64;
constexpr int Bb = 2, Ss = 1024, NT = Bb * Ss;   // 2048 tokens

// ---------------------------------------------------------------------------
// fragment loaders (layouts per CDNA5 ISA 7.12.2, wave32)
// A-frag (16x32 f16): lane lr = row M; elems e<8 -> K = half*8+e, e>=8 -> K = 16+half*8+(e-8)
// B-frag (32x16 f16): lane lr = col N; elems e<8 -> K = half*8+e, e>=8 -> K = 16+half*8+(e-8)
// C/D   (16x16 f32): lane lr = col N; vgpr v -> row M = v + half*8
// ---------------------------------------------------------------------------
__device__ __forceinline__ v16h mk16(h8 lo, h8 hi) {
  v16h r;
#pragma unroll
  for (int i = 0; i < 8; ++i) { r[i] = lo[i]; r[i + 8] = hi[i]; }
  return r;
}
// base must already include +half*8; chunks at +0 and +16
__device__ __forceinline__ v16h load_fragA(const _Float16* base) {
  return mk16(*(const h8*)(base), *(const h8*)(base + 16));
}
// base must already include +half*8; chunks at +0 and +16
__device__ __forceinline__ v16h load_fragB(const _Float16* base) {
  return mk16(*(const h8*)(base), *(const h8*)(base + 16));
}
#define WMMA_F16(A_, B_, C_) \
  __builtin_amdgcn_wmma_f32_16x16x32_f16(false, (A_), false, (B_), (short)0, (C_), false, false)

// ---------------------------------------------------------------------------
// group-wise ternary quantize+dequant: f32 -> f16, groups of 128 (1 wave/group)
// ---------------------------------------------------------------------------
__global__ __launch_bounds__(256) void quantize_groups(
    const float* __restrict__ W, _Float16* __restrict__ Wq, long n) {
  long g = (long)blockIdx.x * 8 + (threadIdx.x >> 5);
  int lane = threadIdx.x & 31;
  long base = g * GSz + lane * 4;
  float4 w = *(const float4*)(W + base);
  float am = fmaxf(fmaxf(fabsf(w.x), fabsf(w.y)), fmaxf(fabsf(w.z), fabsf(w.w)));
#pragma unroll
  for (int m = 16; m >= 1; m >>= 1) am = fmaxf(am, __shfl_xor(am, m, 32));
  float s = fmaxf(am, 1e-8f), inv = 1.0f / s;
  h4 o;
  o[0] = (_Float16)(fminf(fmaxf(rintf(w.x * inv), -1.f), 1.f) * s);
  o[1] = (_Float16)(fminf(fmaxf(rintf(w.y * inv), -1.f), 1.f) * s);
  o[2] = (_Float16)(fminf(fmaxf(rintf(w.z * inv), -1.f), 1.f) * s);
  o[3] = (_Float16)(fminf(fmaxf(rintf(w.w * inv), -1.f), 1.f) * s);
  *(h4*)(Wq + base) = o;
}

// ---------------------------------------------------------------------------
// embedding: dequantize-on-the-fly gather (D%128==0 -> groups align to rows)
// ---------------------------------------------------------------------------
__global__ __launch_bounds__(256) void embed_gather(
    const float* __restrict__ emb, const int* __restrict__ tok, float* __restrict__ X) {
  int row = blockIdx.x;
  int t = tok[row];
  int w = threadIdx.x >> 5, lane = threadIdx.x & 31;
  long off = (long)t * Dm + w * GSz + lane * 4;
  float4 v = *(const float4*)(emb + off);
  float am = fmaxf(fmaxf(fabsf(v.x), fabsf(v.y)), fmaxf(fabsf(v.z), fabsf(v.w)));
#pragma unroll
  for (int m = 16; m >= 1; m >>= 1) am = fmaxf(am, __shfl_xor(am, m, 32));
  float s = fmaxf(am, 1e-8f), inv = 1.0f / s;
  float4 o;
  o.x = fminf(fmaxf(rintf(v.x * inv), -1.f), 1.f) * s;
  o.y = fminf(fmaxf(rintf(v.y * inv), -1.f), 1.f) * s;
  o.z = fminf(fmaxf(rintf(v.z * inv), -1.f), 1.f) * s;
  o.w = fminf(fmaxf(rintf(v.w * inv), -1.f), 1.f) * s;
  *(float4*)(X + (long)row * Dm + w * GSz + lane * 4) = o;
}

// ---------------------------------------------------------------------------
// RMSNorm: f32 row -> f16 row (GEMM input precision)
// ---------------------------------------------------------------------------
__global__ __launch_bounds__(256) void rmsnorm_f16(
    const float* __restrict__ X, const float* __restrict__ g, _Float16* __restrict__ H) {
  int row = blockIdx.x, t = threadIdx.x;
  int w = t >> 5, lane = t & 31;
  float4 v = *(const float4*)(X + (long)row * Dm + t * 4);
  float ss = v.x * v.x + v.y * v.y + v.z * v.z + v.w * v.w;
#pragma unroll
  for (int m = 16; m >= 1; m >>= 1) ss += __shfl_xor(ss, m, 32);
  __shared__ float part[8];
  __shared__ float sinv;
  if (lane == 0) part[w] = ss;
  __syncthreads();
  if (t == 0) {
    float tot = 0.f;
#pragma unroll
    for (int i = 0; i < 8; ++i) tot += part[i];
    sinv = rsqrtf(tot * (1.0f / Dm) + 1e-6f);
  }
  __syncthreads();
  float iv = sinv;
  float4 gv = *(const float4*)(g + t * 4);
  h4 o;
  o[0] = (_Float16)(v.x * iv * gv.x);
  o[1] = (_Float16)(v.y * iv * gv.y);
  o[2] = (_Float16)(v.z * iv * gv.z);
  o[3] = (_Float16)(v.w * iv * gv.w);
  *(h4*)(H + (long)row * Dm + t * 4) = o;
}

// ---------------------------------------------------------------------------
// WMMA GEMM: C[M,N] = A[M,K] * W[N,K]^T   (F.linear)
// block = 256 thr = 8 waves; block tile 128x64; wave tile 32x32 (2x2 frags)
// K unrolled x2 (K % 64 == 0): 8 v_wmma per iteration; weight stream
// prefetched 512B ahead (global_prefetch_b8).
// MODE 0: store f32 | 1: store f16 | 2: f32 residual add (Aux=res, may alias)
// MODE 3: f16 silu(Aux)*acc (SwiGLU up-proj fuse, Aux=gate f32)
// ---------------------------------------------------------------------------
template <int MODE>
__global__ __launch_bounds__(256) void gemm_wmma(
    const _Float16* __restrict__ A, const _Float16* __restrict__ W,
    void* Out, const float* Aux, int M, int N, int K) {
  int wave = threadIdx.x >> 5, lane = threadIdx.x & 31;
  int lr = lane & 15, half = lane >> 4;
  int wm = wave & 3, wn = wave >> 2;
  int rowBase = blockIdx.y * 128 + wm * 32;
  int colBase = blockIdx.x * 64 + wn * 32;

  const _Float16* a0p = A + (long)(rowBase + lr) * K + half * 8;
  const _Float16* a1p = a0p + (long)16 * K;
  const _Float16* b0p = W + (long)(colBase + lr) * K + half * 8;
  const _Float16* b1p = b0p + (long)16 * K;

  v8f acc[2][2] = {};
  for (int k0 = 0; k0 < K; k0 += 64) {
    // prefetch the streaming weight rows ~512B ahead
    __builtin_prefetch(b0p + k0 + 256, 0, 1);
    __builtin_prefetch(b1p + k0 + 256, 0, 1);
#pragma unroll
    for (int ku = 0; ku < 2; ++ku) {
      int k = k0 + ku * 32;
      v16h a[2] = { load_fragA(a0p + k), load_fragA(a1p + k) };
      v16h b[2] = { load_fragB(b0p + k), load_fragB(b1p + k) };
#pragma unroll
      for (int i = 0; i < 2; ++i)
#pragma unroll
        for (int j = 0; j < 2; ++j) acc[i][j] = WMMA_F16(a[i], b[j], acc[i][j]);
    }
  }

#pragma unroll
  for (int i = 0; i < 2; ++i)
#pragma unroll
    for (int j = 0; j < 2; ++j) {
      int col = colBase + j * 16 + lr;
#pragma unroll
      for (int v = 0; v < 8; ++v) {
        int row = rowBase + i * 16 + v + half * 8;
        long idx = (long)row * N + col;
        float val = acc[i][j][v];
        if (MODE == 0) {
          ((float*)Out)[idx] = val;
        } else if (MODE == 1) {
          ((_Float16*)Out)[idx] = (_Float16)val;
        } else if (MODE == 2) {
          ((float*)Out)[idx] = Aux[idx] + val;
        } else {
          float gt = Aux[idx];
          float sg = gt / (1.0f + expf(-gt));
          ((_Float16*)Out)[idx] = (_Float16)(sg * val);
        }
      }
    }
}

// ---------------------------------------------------------------------------
// V transpose: Vt[bh][d][s] = V[(b*S+s)*D + h*64 + d]  (coalesced writes)
// ---------------------------------------------------------------------------
__global__ __launch_bounds__(256) void transpose_v(
    const _Float16* __restrict__ Vb, _Float16* __restrict__ Vt) {
  long o = (long)blockIdx.x * 256 + threadIdx.x;
  int s = (int)(o & (Ss - 1));
  int d = (int)((o >> 10) & (DHd - 1));
  int bh = (int)(o >> 16);
  int b = bh >> 4, h = bh & (Hh - 1);
  Vt[o] = Vb[((long)(b * Ss + s)) * Dm + h * DHd + d];
}

// ---------------------------------------------------------------------------
// Flash attention, 1 wave per (bh, 16-row block); causal, online softmax.
// Epilogue fuses per-head residual mix: out = softmax(QK^T/8)V + alpha[h]*h
// ---------------------------------------------------------------------------
__global__ __launch_bounds__(32) void attn_flash(
    const _Float16* __restrict__ Q, const _Float16* __restrict__ Kb,
    const _Float16* __restrict__ Vt, const _Float16* __restrict__ Hn,
    const float* __restrict__ alpha, _Float16* __restrict__ Om) {
  int r0 = blockIdx.x * 16;
  int bh = blockIdx.y;
  int b = bh >> 4, h = bh & (Hh - 1);
  int lane = threadIdx.x & 31;
  int lr = lane & 15, half = lane >> 4;

  const _Float16* qrow = Q + ((long)(b * Ss + r0 + lr)) * Dm + h * DHd + half * 8;
  v16h qa[2] = { load_fragA(qrow), load_fragA(qrow + 32) };

  float mrow[8], lrow[8];
  v8f oacc[4] = {};
#pragma unroll
  for (int v = 0; v < 8; ++v) { mrow[v] = -1e30f; lrow[v] = 0.f; }

  __shared__ __align__(16) _Float16 pl[16][40];
  const float scale = 0.125f;  // DH^-0.5

  for (int t0 = 0; t0 < r0 + 16; t0 += 32) {
    v8f sc[2] = {};
#pragma unroll
    for (int j = 0; j < 2; ++j) {
      const _Float16* krow =
          Kb + ((long)(b * Ss + t0 + j * 16 + lr)) * Dm + h * DHd + half * 8;
      sc[j] = WMMA_F16(qa[0], load_fragB(krow), sc[j]);
      sc[j] = WMMA_F16(qa[1], load_fragB(krow + 32), sc[j]);
    }
#pragma unroll
    for (int j = 0; j < 2; ++j) {
      int t = t0 + j * 16 + lr;
#pragma unroll
      for (int v = 0; v < 8; ++v) {
        int r = r0 + v + half * 8;
        float x = sc[j][v] * scale;
        sc[j][v] = (t > r) ? -1e30f : x;
      }
    }
    float p0[8], p1[8];
#pragma unroll
    for (int v = 0; v < 8; ++v) {
      float mx = fmaxf(sc[0][v], sc[1][v]);
#pragma unroll
      for (int m = 8; m >= 1; m >>= 1) mx = fmaxf(mx, __shfl_xor(mx, m, 32));
      float mn = fmaxf(mrow[v], mx);
      float c = expf(mrow[v] - mn);
      float e0 = expf(sc[0][v] - mn);
      float e1 = expf(sc[1][v] - mn);
      float ps = e0 + e1;
#pragma unroll
      for (int m = 8; m >= 1; m >>= 1) ps += __shfl_xor(ps, m, 32);
      lrow[v] = lrow[v] * c + ps;
      mrow[v] = mn;
      p0[v] = e0; p1[v] = e1;
#pragma unroll
      for (int d = 0; d < 4; ++d) oacc[d][v] *= c;
    }
    // C-layout -> A-layout via LDS
#pragma unroll
    for (int v = 0; v < 8; ++v) {
      pl[v + half * 8][lr] = (_Float16)p0[v];
      pl[v + half * 8][16 + lr] = (_Float16)p1[v];
    }
    __syncthreads();
    v16h pa = mk16(*(const h8*)&pl[lr][half * 8], *(const h8*)&pl[lr][16 + half * 8]);
    __syncthreads();
#pragma unroll
    for (int d = 0; d < 4; ++d) {
      const _Float16* vrow =
          Vt + ((long)bh * DHd + d * 16 + lr) * Ss + t0 + half * 8;
      oacc[d] = WMMA_F16(pa, load_fragB(vrow), oacc[d]);
    }
  }

  float av = alpha[h];
#pragma unroll
  for (int d = 0; d < 4; ++d) {
    int dd = d * 16 + lr;
#pragma unroll
    for (int v = 0; v < 8; ++v) {
      int r = r0 + v + half * 8;
      long idx = ((long)(b * Ss + r)) * Dm + h * DHd + dd;
      float o = oacc[d][v] / lrow[v];
      Om[idx] = (_Float16)(o + av * (float)Hn[idx]);
    }
  }
}

// ---------------------------------------------------------------------------
extern "C" void kernel_launch(void* const* d_in, const int* in_sizes, int n_in,
                              void* d_out, int out_size, void* d_ws, size_t ws_size,
                              hipStream_t stream) {
  (void)n_in; (void)out_size; (void)ws_size;
  const int* tokens; const float *emb, *na, *nm, *wq, *wk, *wv, *wo, *al, *wg, *wu, *wd, *nf, *hd;
  if (in_sizes[0] == NT) {  // insertion-order flatten
    tokens = (const int*)d_in[0];  emb = (const float*)d_in[1];
    na = (const float*)d_in[2];    nm = (const float*)d_in[3];
    wq = (const float*)d_in[4];    wk = (const float*)d_in[5];
    wv = (const float*)d_in[6];    wo = (const float*)d_in[7];
    al = (const float*)d_in[8];    wg = (const float*)d_in[9];
    wu = (const float*)d_in[10];   wd = (const float*)d_in[11];
    nf = (const float*)d_in[12];   hd = (const float*)d_in[13];
  } else {                  // sorted pytree flatten
    emb = (const float*)d_in[0];   hd = (const float*)d_in[1];
    al = (const float*)d_in[2];    na = (const float*)d_in[3];
    nm = (const float*)d_in[4];    wd = (const float*)d_in[5];
    wg = (const float*)d_in[6];    wk = (const float*)d_in[7];
    wo = (const float*)d_in[8];    wq = (const float*)d_in[9];
    wu = (const float*)d_in[10];   wv = (const float*)d_in[11];
    nf = (const float*)d_in[12];   tokens = (const int*)d_in[13];
  }

  char* ws = (char*)d_ws;
  size_t off = 0;
  auto alloc = [&](size_t bytes) -> void* {
    void* p = ws + off;
    off = (off + bytes + 255) & ~(size_t)255;
    return p;
  };

  const long nQKVO = (long)Ll * Dm * Dm;
  const long nGU   = (long)Ll * FFf * Dm;
  const long nHead = (long)Vv * Dm;

  _Float16* wq_q = (_Float16*)alloc(nQKVO * 2);
  _Float16* wk_q = (_Float16*)alloc(nQKVO * 2);
  _Float16* wv_q = (_Float16*)alloc(nQKVO * 2);
  _Float16* wo_q = (_Float16*)alloc(nQKVO * 2);
  _Float16* wg_q = (_Float16*)alloc(nGU * 2);
  _Float16* wu_q = (_Float16*)alloc(nGU * 2);
  _Float16* wd_q = (_Float16*)alloc(nGU * 2);
  _Float16* hd_q = (_Float16*)alloc(nHead * 2);
  float*    X    = (float*)   alloc((long)NT * Dm * 4);
  _Float16* Hb   = (_Float16*)alloc((long)NT * Dm * 2);
  _Float16* Qb   = (_Float16*)alloc((long)NT * Dm * 2);
  _Float16* Kb   = (_Float16*)alloc((long)NT * Dm * 2);
  _Float16* Vb   = (_Float16*)alloc((long)NT * Dm * 2);
  _Float16* Vt   = (_Float16*)alloc((long)NT * Dm * 2);
  _Float16* Om   = (_Float16*)alloc((long)NT * Dm * 2);
  float*    Gf   = (float*)   alloc((long)NT * FFf * 4);
  _Float16* Act  = (_Float16*)alloc((long)NT * FFf * 2);

  auto quant = [&](const float* s, _Float16* dst, long n) {
    quantize_groups<<<dim3((unsigned)(n / GSz / 8)), 256, 0, stream>>>(s, dst, n);
  };
  quant(wq, wq_q, nQKVO); quant(wk, wk_q, nQKVO);
  quant(wv, wv_q, nQKVO); quant(wo, wo_q, nQKVO);
  quant(wg, wg_q, nGU);   quant(wu, wu_q, nGU);   quant(wd, wd_q, nGU);
  quant(hd, hd_q, nHead);

  embed_gather<<<dim3(NT), 256, 0, stream>>>(emb, tokens, X);

  for (int l = 0; l < Ll; ++l) {
    const _Float16* Wq_l = wq_q + (long)l * Dm * Dm;
    const _Float16* Wk_l = wk_q + (long)l * Dm * Dm;
    const _Float16* Wv_l = wv_q + (long)l * Dm * Dm;
    const _Float16* Wo_l = wo_q + (long)l * Dm * Dm;
    const _Float16* Wg_l = wg_q + (long)l * FFf * Dm;
    const _Float16* Wu_l = wu_q + (long)l * FFf * Dm;
    const _Float16* Wd_l = wd_q + (long)l * Dm * FFf;

    rmsnorm_f16<<<dim3(NT), 256, 0, stream>>>(X, na + (long)l * Dm, Hb);
    dim3 gD(Dm / 64, NT / 128), gF(FFf / 64, NT / 128);
    gemm_wmma<1><<<gD, 256, 0, stream>>>(Hb, Wq_l, Qb, nullptr, NT, Dm, Dm);
    gemm_wmma<1><<<gD, 256, 0, stream>>>(Hb, Wk_l, Kb, nullptr, NT, Dm, Dm);
    gemm_wmma<1><<<gD, 256, 0, stream>>>(Hb, Wv_l, Vb, nullptr, NT, Dm, Dm);
    transpose_v<<<dim3((unsigned)(((long)NT * Dm) / 256)), 256, 0, stream>>>(Vb, Vt);
    attn_flash<<<dim3(Ss / 16, Bb * Hh), 32, 0, stream>>>(
        Qb, Kb, Vt, Hb, al + (long)l * Hh, Om);
    gemm_wmma<2><<<gD, 256, 0, stream>>>(Om, Wo_l, X, X, NT, Dm, Dm);

    rmsnorm_f16<<<dim3(NT), 256, 0, stream>>>(X, nm + (long)l * Dm, Hb);
    gemm_wmma<0><<<gF, 256, 0, stream>>>(Hb, Wg_l, Gf, nullptr, NT, FFf, Dm);
    gemm_wmma<3><<<gF, 256, 0, stream>>>(Hb, Wu_l, Act, Gf, NT, FFf, Dm);
    gemm_wmma<2><<<gD, 256, 0, stream>>>(Act, Wd_l, X, X, NT, Dm, FFf);
  }

  rmsnorm_f16<<<dim3(NT), 256, 0, stream>>>(X, nf, Hb);
  gemm_wmma<0><<<dim3(Vv / 64, NT / 128), 256, 0, stream>>>(
      Hb, hd_q, (float*)d_out, nullptr, NT, Vv, Dm);
}
